// SingleRNN_81561428951129
// MI455X (gfx1250) — matrix operand order repaired and checked by reading, not compiled, not run on hardware
//
#include <hip/hip_runtime.h>
#include <hip/hip_bf16.h>
#include <cstdint>

// ---------- types ----------
typedef _Float16 v16h __attribute__((ext_vector_type(16)));
typedef _Float16 h8   __attribute__((ext_vector_type(8)));
typedef float    v8f  __attribute__((ext_vector_type(8)));
typedef float    f8   __attribute__((ext_vector_type(8)));

#define HDIM 512
#define G3   1536      // 3*H
#define NB   128       // batch * dilation
#define TE   256       // encoder steps
#define TD   48        // decoder steps that matter (t=48 is truncated away)
#define FH   96

static __device__ __forceinline__ v8f wmma_f16(v16h a, v16h b, v8f c) {
  // D = A(16x32 f16) * B(32x16 f16) + C(16x16 f32)
  return __builtin_amdgcn_wmma_f32_16x16x32_f16(false, a, false, b, (short)0, c,
                                                false, false);
}

static __device__ __forceinline__ float sigf(float x) {
  return 1.0f / (1.0f + __expf(-x));
}
static __device__ __forceinline__ float tanh_fast(float x) {
  return 2.0f / (1.0f + __expf(-2.0f * x)) - 1.0f;
}

// A fragment (16x32, f16) from row-major f16 matrix.
// ISA layout: lanes 0-15 hold M=lane, K = {kb+0..7, kb+16..23};
//             lanes 16-31 hold M=lane-16, K = {kb+8..15, kb+24..31}.
static __device__ __forceinline__ v16h load_a_f16(const _Float16* __restrict__ A,
                                                  int ld, int kbase, int lane) {
  int m = lane & 15, kh = lane >> 4;
  const _Float16* p = A + (size_t)m * ld + kbase + kh * 8;
  h8 lo = *(const h8*)(p);
  h8 hi = *(const h8*)(p + 16);
  return __builtin_shufflevector(lo, hi, 0,1,2,3,4,5,6,7,8,9,10,11,12,13,14,15);
}

// A fragment from row-major f32 matrix (recurrent h), converted to f16.
static __device__ __forceinline__ v16h load_a_f32(const float* __restrict__ A,
                                                  int ld, int kbase, int lane) {
  int m = lane & 15, kh = lane >> 4;
  const float* p = A + (size_t)m * ld + kbase + kh * 8;
  f8 lo = *(const f8*)(p);
  f8 hi = *(const f8*)(p + 16);
  v16h a;
#pragma unroll
  for (int e = 0; e < 8; ++e) {
    a[e]     = (_Float16)lo[e];
    a[8 + e] = (_Float16)hi[e];
  }
  return a;
}

// B fragment (32x16, f16) from pre-packed global weights.
static __device__ __forceinline__ v16h load_b_pack(const _Float16* __restrict__ pack,
                                                   int ntile, int KT, int kt, int lane) {
  return *(const v16h*)(pack + (((size_t)ntile * KT + kt) * 32 + lane) * 16);
}

// B fragment from LDS-staged region (KT=16 fixed; region r3 = 8192 f16).
static __device__ __forceinline__ v16h lds_b_frag(const _Float16* sB, int r3, int kt, int lane) {
  return *(const v16h*)(sB + (((r3 * 16) + kt) * 32 + lane) * 16);
}

// Async-stage 3 B-tile regions (ntiles c, c+32, c+64; KT=16 -> 16KB each) of a
// pack into LDS using gfx1250 GLOBAL_LOAD_ASYNC_TO_LDS_B128 (ASYNCcnt-tracked).
static __device__ __forceinline__ void stage_pack3(const _Float16* __restrict__ pack,
                                                   _Float16* sB, int c, int tid) {
#pragma unroll
  for (int r3 = 0; r3 < 3; ++r3) {
    const char* gsrc = (const char*)(pack + (size_t)(c + 32 * r3) * 8192);
    uint32_t lbase = (uint32_t)(uintptr_t)(sB + r3 * 8192);
#pragma unroll
    for (int q = 0; q < 4; ++q) {           // 16384 B / (256 thr * 16 B)
      int byteoff = (q * 256 + tid) * 16;
      uint32_t laddr = lbase + (uint32_t)byteoff;
      uint64_t gaddr = (uint64_t)(uintptr_t)(gsrc + byteoff);
      asm volatile("global_load_async_to_lds_b128 %0, %1, off"
                   :: "v"(laddr), "v"(gaddr) : "memory");
    }
  }
}

static __device__ __forceinline__ void wait_async_and_barrier() {
  asm volatile("s_wait_asynccnt 0x0" ::: "memory");
  __syncthreads();
}

// ---------- pack W[n][k] (f32, row length ldW) into B-fragment layout ----------
// pack[(nt*KT+kt)*32 + lane][e] = W[nt*16 + (lane&15)][kOff + kt*32 + (lane>>4)*8 + (e<8 ? e : e+8)]
__global__ void __launch_bounds__(32)
pack_weight(const float* __restrict__ W, int ldW, int kOff, int KT,
            _Float16* __restrict__ pack) {
  int blk = blockIdx.x;              // nt*KT + kt
  int nt = blk / KT, kt = blk % KT;
  int l = threadIdx.x;
  const float* src = W + (size_t)(nt * 16 + (l & 15)) * ldW + kOff + kt * 32 + (l >> 4) * 8;
  _Float16* dst = pack + ((size_t)blk * 32 + l) * 16;
#pragma unroll
  for (int e = 0; e < 16; ++e) dst[e] = (_Float16)src[e < 8 ? e : e + 8];
}

// ---------- dilation gather: X0(64,1024,512) -> Xenc f16 (32768 x 1024), row m=t*128+n ----------
__global__ void build_xenc(const float* __restrict__ X0, _Float16* __restrict__ Xe) {
  size_t id = (size_t)blockIdx.x * blockDim.x + threadIdx.x; // < 32768*1024
  int c = (int)(id & 1023);
  int m = (int)(id >> 10);
  int t = m >> 7, n = m & 127;
  int d = n >> 6, b = n & 63;
  Xe[id] = (_Float16)X0[((size_t)b * 1024 + c) * 512 + 2 * t + d];
}

// ---------- X1(64,512,96) -> Xdec f16 (6144 x 512), row m=t*128+n, t<48 ----------
__global__ void build_xdec(const float* __restrict__ X1, _Float16* __restrict__ Xd) {
  size_t id = (size_t)blockIdx.x * blockDim.x + threadIdx.x; // < 6144*512
  int c = (int)(id & 511);
  int m = (int)(id >> 9);
  int t = m >> 7, n = m & 127;
  int d = n >> 6, b = n & 63;
  Xd[id] = (_Float16)X1[((size_t)b * 512 + c) * 96 + 2 * t + d];
}

__global__ void zero_f32(float* __restrict__ p, int n) {
  int i = blockIdx.x * 256 + threadIdx.x;
  if (i < n) p[i] = 0.0f;
}

// ---------- big pre-GEMM: Out(f16, M x 1536) = A(f16, M x K) @ Bpack + bias ----------
// one wave per 32x64 output tile: 8 WMMA per K-tile. (B-pack is L2-resident.)
__global__ void __launch_bounds__(32)
gemm_bias_f16(const _Float16* __restrict__ A, int K, int KT,
              const _Float16* __restrict__ Bpack,
              const float* __restrict__ bias,
              _Float16* __restrict__ Out) {
  int rt = blockIdx.x;   // 32-row tile
  int cg = blockIdx.y;   // 64-col group (4 ntiles)
  int l = threadIdx.x;
  v8f zero = {0.f, 0.f, 0.f, 0.f, 0.f, 0.f, 0.f, 0.f};
  v8f acc[2][4];
#pragma unroll
  for (int s = 0; s < 2; ++s)
#pragma unroll
    for (int n4 = 0; n4 < 4; ++n4) acc[s][n4] = zero;

  const _Float16* A0 = A + (size_t)(rt * 32) * K;
  const _Float16* A1 = A0 + (size_t)16 * K;
  for (int kt = 0; kt < KT; ++kt) {
    v16h a0 = load_a_f16(A0, K, kt * 32, l);
    v16h a1 = load_a_f16(A1, K, kt * 32, l);
#pragma unroll
    for (int n4 = 0; n4 < 4; ++n4) {
      v16h b = load_b_pack(Bpack, cg * 4 + n4, KT, kt, l);
      acc[0][n4] = wmma_f16(a0, b, acc[0][n4]);
      acc[1][n4] = wmma_f16(a1, b, acc[1][n4]);
    }
  }
#pragma unroll
  for (int s = 0; s < 2; ++s)
#pragma unroll
    for (int n4 = 0; n4 < 4; ++n4) {
      int j = (cg * 4 + n4) * 16 + (l & 15);
      float bj = bias[j];
#pragma unroll
      for (int v = 0; v < 8; ++v) {
        int m = rt * 32 + s * 16 + (l >> 4) * 8 + v;
        Out[(size_t)m * G3 + j] = (_Float16)(acc[s][n4][v] + bj);
      }
    }
}

// ---------- encoder recurrent step: gh = h @ Whh.T, fused GRU gates ----------
// One 256-thread block (8 row-tile waves) per 16-col tile; r/z/n B tiles
// async-staged to LDS once per block (8x less L2 traffic than per-wave loads).
__global__ void __launch_bounds__(256)
enc_step(const float* __restrict__ hCur, float* __restrict__ hNxt,
         const _Float16* __restrict__ GiAll, int t,
         const _Float16* __restrict__ WhhPack, const float* __restrict__ bhh) {
  __shared__ _Float16 sB[3 * 8192];   // 48 KB: r/z/n tiles, full K=512
  int tid = threadIdx.x;
  int l = tid & 31;
  int rt = tid >> 5;     // 0..7 (16 batch rows per wave)
  int c = blockIdx.x;    // 0..31 (16 hidden cols)

  stage_pack3(WhhPack, sB, c, tid);
  wait_async_and_barrier();

  v8f ar = {0.f,0.f,0.f,0.f,0.f,0.f,0.f,0.f};
  v8f az = ar, an = ar;
  const float* A0 = hCur + (size_t)(rt * 16) * HDIM;
  for (int kt = 0; kt < 16; ++kt) {
    v16h a = load_a_f32(A0, HDIM, kt * 32, l);
    ar = wmma_f16(a, lds_b_frag(sB, 0, kt, l), ar);
    az = wmma_f16(a, lds_b_frag(sB, 1, kt, l), az);
    an = wmma_f16(a, lds_b_frag(sB, 2, kt, l), an);
  }
  int j = c * 16 + (l & 15);
  float bhr = bhh[j], bhz = bhh[HDIM + j], bhn = bhh[2 * HDIM + j];
  const _Float16* gi = GiAll + (size_t)t * NB * G3;   // b_ih already folded in
#pragma unroll
  for (int v = 0; v < 8; ++v) {
    int m = rt * 16 + (l >> 4) * 8 + v;
    size_t g = (size_t)m * G3 + j;
    float rr = sigf((float)gi[g] + ar[v] + bhr);
    float zz = sigf((float)gi[g + HDIM] + az[v] + bhz);
    float nn = tanh_fast((float)gi[g + 2 * HDIM] + rr * (an[v] + bhn));
    float ho = hCur[(size_t)m * HDIM + j];
    hNxt[(size_t)m * HDIM + j] = (1.0f - zz) * nn + zz * ho;
  }
}

// ---------- decoder recurrent step ----------
// x = concat(h, x1_t): gi = h @ WihL.T + Gi1[t] (x1 part + b_ih prefolded)
//                      gh = h @ Whh.T + b_hh.
// Two-phase LDS staging (same 48 KB buffer) to keep static LDS conservative.
__global__ void __launch_bounds__(256)
dec_step(const float* __restrict__ hCur, float* __restrict__ hNxt,
         const _Float16* __restrict__ Gi1All, int t,
         const _Float16* __restrict__ WihLPack, const _Float16* __restrict__ WhhPack,
         const float* __restrict__ bhh, float* __restrict__ out) {
  __shared__ _Float16 sB[3 * 8192];
  int tid = threadIdx.x;
  int l = tid & 31;
  int rt = tid >> 5;
  int c = blockIdx.x;
  const float* A0 = hCur + (size_t)(rt * 16) * HDIM;
  v8f z8 = {0.f,0.f,0.f,0.f,0.f,0.f,0.f,0.f};

  // phase 1: gi contributions from h @ WihL.T
  stage_pack3(WihLPack, sB, c, tid);
  wait_async_and_barrier();
  v8f gr = z8, gz = z8, gn = z8;
  for (int kt = 0; kt < 16; ++kt) {
    v16h a = load_a_f32(A0, HDIM, kt * 32, l);
    gr = wmma_f16(a, lds_b_frag(sB, 0, kt, l), gr);
    gz = wmma_f16(a, lds_b_frag(sB, 1, kt, l), gz);
    gn = wmma_f16(a, lds_b_frag(sB, 2, kt, l), gn);
  }
  __syncthreads();   // everyone done reading phase-1 LDS before restage

  // phase 2: gh = h @ Whh.T
  stage_pack3(WhhPack, sB, c, tid);
  wait_async_and_barrier();
  v8f hr = z8, hz = z8, hn = z8;
  for (int kt = 0; kt < 16; ++kt) {
    v16h a = load_a_f32(A0, HDIM, kt * 32, l);
    hr = wmma_f16(a, lds_b_frag(sB, 0, kt, l), hr);
    hz = wmma_f16(a, lds_b_frag(sB, 1, kt, l), hz);
    hn = wmma_f16(a, lds_b_frag(sB, 2, kt, l), hn);
  }

  int j = c * 16 + (l & 15);
  float bhr = bhh[j], bhz = bhh[HDIM + j], bhn = bhh[2 * HDIM + j];
  const _Float16* gi1 = Gi1All + (size_t)t * NB * G3;
#pragma unroll
  for (int v = 0; v < 8; ++v) {
    int m = rt * 16 + (l >> 4) * 8 + v;       // batch row n
    size_t g = (size_t)m * G3 + j;
    float rr = sigf((float)gi1[g] + gr[v] + hr[v] + bhr);
    float zz = sigf((float)gi1[g + HDIM] + gz[v] + hz[v] + bhz);
    float nn = tanh_fast((float)gi1[g + 2 * HDIM] + gn[v] + rr * (hn[v] + bhn));
    float ho = hCur[(size_t)m * HDIM + j];
    float h2 = (1.0f - zz) * nn + zz * ho;
    hNxt[(size_t)m * HDIM + j] = h2;
    int d = m >> 6, b = m & 63;
    int f = 2 * t + d;                         // < 96 for t < 48
    out[((size_t)b * HDIM + j) * FH + f] = h2;
  }
}

extern "C" void kernel_launch(void* const* d_in, const int* in_sizes, int n_in,
                              void* d_out, int out_size, void* d_ws, size_t ws_size,
                              hipStream_t stream) {
  (void)in_sizes; (void)n_in; (void)out_size; (void)ws_size;
  const float* X0   = (const float*)d_in[0];
  const float* X1   = (const float*)d_in[1];
  // d_in[2] = seqlen (fixed 512, unused)
  const float* WihE = (const float*)d_in[3];
  const float* WhhE = (const float*)d_in[4];
  const float* bihE = (const float*)d_in[5];
  const float* bhhE = (const float*)d_in[6];
  const float* WihD = (const float*)d_in[7];
  const float* WhhD = (const float*)d_in[8];
  const float* bihD = (const float*)d_in[9];
  const float* bhhD = (const float*)d_in[10];
  float* out = (float*)d_out;

  uint8_t* ws = (uint8_t*)d_ws;
  size_t off = 0;
  auto walloc = [&](size_t bytes) -> void* {
    void* p = ws + off;
    off = (off + bytes + 255) & ~(size_t)255;
    return p;
  };
  _Float16* pWihE  = (_Float16*)walloc((size_t)96 * 32 * 512 * 2); // K=1024
  _Float16* pWhhE  = (_Float16*)walloc((size_t)96 * 16 * 512 * 2); // K=512
  _Float16* pWihDL = (_Float16*)walloc((size_t)96 * 16 * 512 * 2);
  _Float16* pWihDR = (_Float16*)walloc((size_t)96 * 16 * 512 * 2);
  _Float16* pWhhD  = (_Float16*)walloc((size_t)96 * 16 * 512 * 2);
  _Float16* Xe  = (_Float16*)walloc((size_t)TE * NB * 1024 * 2);   // 64 MB
  _Float16* GiE = (_Float16*)walloc((size_t)TE * NB * G3 * 2);     // 96 MB
  _Float16* Xd  = (_Float16*)walloc((size_t)TD * NB * 512 * 2);
  _Float16* GiD = (_Float16*)walloc((size_t)TD * NB * G3 * 2);
  float* hA = (float*)walloc((size_t)NB * HDIM * 4);
  float* hB = (float*)walloc((size_t)NB * HDIM * 4);

  // weight packing into WMMA B-fragment layout
  pack_weight<<<96 * 32, 32, 0, stream>>>(WihE, 1024, 0,   32, pWihE);
  pack_weight<<<96 * 16, 32, 0, stream>>>(WhhE, 512,  0,   16, pWhhE);
  pack_weight<<<96 * 16, 32, 0, stream>>>(WihD, 1024, 0,   16, pWihDL);
  pack_weight<<<96 * 16, 32, 0, stream>>>(WihD, 1024, 512, 16, pWihDR);
  pack_weight<<<96 * 16, 32, 0, stream>>>(WhhD, 512,  0,   16, pWhhD);

  // dilation gathers (f32 -> f16)
  build_xenc<<<(int)(((size_t)TE * NB * 1024) / 256), 256, 0, stream>>>(X0, Xe);
  build_xdec<<<(int)(((size_t)TD * NB * 512) / 256), 256, 0, stream>>>(X1, Xd);

  // time-parallel pre-GEMMs (b_ih folded in)
  gemm_bias_f16<<<dim3(TE * NB / 32, 24), 32, 0, stream>>>(Xe, 1024, 32, pWihE, bihE, GiE);
  gemm_bias_f16<<<dim3(TD * NB / 32, 24), 32, 0, stream>>>(Xd, 512, 16, pWihDR, bihD, GiD);

  // h0 = 0
  zero_f32<<<(NB * HDIM) / 256, 256, 0, stream>>>(hA, NB * HDIM);

  float* cur = hA;
  float* nxt = hB;
  for (int t = 0; t < TE; ++t) {
    enc_step<<<32, 256, 0, stream>>>(cur, nxt, GiE, t, pWhhE, bhhE);
    float* tmp = cur; cur = nxt; nxt = tmp;
  }
  // decoder starts from h_enc (din == h throughout); t=48 truncated -> skipped
  for (int t = 0; t < TD; ++t) {
    dec_step<<<32, 256, 0, stream>>>(cur, nxt, GiD, t, pWihDL, pWhhD, bhhD, out);
    float* tmp = cur; cur = nxt; nxt = tmp;
  }
}
